// GAT_39444979646796
// MI455X (gfx1250) — matrix operand order repaired and checked by reading, not compiled, not run on hardware
//
#include <hip/hip_runtime.h>

#define NNODES 50000
#define HEADS1 8
#define CH1    32
#define F_IN   64
#define D1     256   // HEADS1*CH1
#define CH2    32

typedef __attribute__((ext_vector_type(16))) _Float16 v16h;
typedef __attribute__((ext_vector_type(8)))  float    v8f;

// ---------------------------------------------------------------------------
// WMMA GEMM: C[M,Nc] = A[M,K] * B[K,Nc]; f32 in (cvt f16), f32 accumulate via
// v_wmma_f32_16x16x32_f16. One wave per 16x32 output tile: one A fragment
// feeds two B fragments / two accumulators (halves A gather traffic, gives
// the scheduler paired back-to-back WMMAs to co-execute with the cvt stream).
// VGPR packing used here (contiguous k, same order in A and B):
//   A (16x32 f16): lanes 0-15 row M=l, halfs K 0..15;
//                  lanes 16-31 row M=l, halfs K 16..31
//   B (32x16 f16): lanes 0-15 col N=l, halfs K 0..15;
//                  lanes 16-31 col N=l, halfs K 16..31
//   C (16x16 f32): VGPR r -> lanes 0-15 M=r, lanes 16-31 M=r+8
// ---------------------------------------------------------------------------
template <int K>
__global__ void wmma_gemm(const float* __restrict__ A, const float* __restrict__ B,
                          float* __restrict__ C, int Nc) {
  const int lane = threadIdx.x;           // 0..31, one wave per block
  const int half = lane >> 4;             // 0 or 1
  const int l    = lane & 15;
  const int row  = blockIdx.x * 16 + l;   // A row this lane supplies
  const int col0 = blockIdx.y * 32 + l;   // first B/C column this lane supplies

  v8f acc0 = {}, acc1 = {};
#pragma unroll 2
  for (int k0 = 0; k0 < K; k0 += 32) {
    v16h a, b0, b1;
    const float* arow = A + (size_t)row * K + k0 + half * 16;
#pragma unroll
    for (int j = 0; j < 8; ++j) {
      a[j]     = (_Float16)arow[j];
      a[8 + j] = (_Float16)arow[8 + j];
    }
    const float* brow = B + (size_t)(k0 + half * 16) * Nc + col0;
#pragma unroll
    for (int j = 0; j < 16; ++j) {
      b0[j] = (_Float16)brow[(size_t)j * Nc];
      b1[j] = (_Float16)brow[(size_t)j * Nc + 16];
    }
    acc0 = __builtin_amdgcn_wmma_f32_16x16x32_f16(false, a, false, b0,
                                                  (short)0, acc0, false, false);
    acc1 = __builtin_amdgcn_wmma_f32_16x16x32_f16(false, a, false, b1,
                                                  (short)0, acc1, false, false);
  }
  const int mbase = blockIdx.x * 16 + half * 8;
  float* crow = C + (size_t)mbase * Nc + col0;
#pragma unroll
  for (int r = 0; r < 8; ++r) {
    crow[(size_t)r * Nc]      = acc0[r];
    crow[(size_t)r * Nc + 16] = acc1[r];
  }
}

// ---------------------------------------------------------------------------
// alpha_src / alpha_dst coefficients: one thread per (node, head)
// ---------------------------------------------------------------------------
template <int H, int C>
__global__ void attn_coeff(const float* __restrict__ h,
                           const float* __restrict__ a_src,
                           const float* __restrict__ a_dst,
                           float* __restrict__ as, float* __restrict__ ad,
                           int total) {
  int idx = blockIdx.x * blockDim.x + threadIdx.x;
  if (idx >= total) return;
  int hh = idx % H;
  const float4* hp = (const float4*)(h + (size_t)idx * C);   // h[n, hh, :]
  const float4* sp = (const float4*)(a_src + (size_t)hh * C);
  const float4* dp = (const float4*)(a_dst + (size_t)hh * C);
  float accs = 0.f, accd = 0.f;
#pragma unroll
  for (int c = 0; c < C / 4; ++c) {
    float4 v = hp[c], s = sp[c], d = dp[c];
    accs += v.x * s.x + v.y * s.y + v.z * s.z + v.w * s.w;
    accd += v.x * d.x + v.y * d.y + v.z * d.z + v.w * d.w;
  }
  as[idx] = accs;
  ad[idx] = accd;
}

__device__ __forceinline__ float leaky(float v) {
  return v >= 0.f ? v : 0.2f * v;
}

// float atomic max via sign-split integer atomics (init pattern 0xFFFFFFFF ok)
__device__ __forceinline__ void atomicMaxF(float* addr, float v) {
  if (v >= 0.f)
    atomicMax((int*)addr, __float_as_int(v));
  else
    atomicMin((unsigned int*)addr, (unsigned int)__float_as_int(v));
}

template <int H>
__device__ __forceinline__ void edge_sd(int idx, const int* src, const int* dst,
                                        int E, int* s, int* d, int* hh) {
  int e = idx / H;             // H is a power of two -> shift
  *hh = idx - e * H;
  if (e < E) { *s = src[e]; *d = dst[e]; }
  else       { *s = e - E;  *d = e - E;  }   // appended self loop
}

template <int H>
__global__ void edge_max(const int* __restrict__ src, const int* __restrict__ dst,
                         int E, const float* __restrict__ as,
                         const float* __restrict__ ad, float* __restrict__ m,
                         int total) {
  int idx = blockIdx.x * blockDim.x + threadIdx.x;
  if (idx >= total) return;
  int s, d, hh;
  edge_sd<H>(idx, src, dst, E, &s, &d, &hh);
  float ev = leaky(as[s * H + hh] + ad[d * H + hh]);
  atomicMaxF(&m[d * H + hh], ev);
}

template <int H>
__global__ void edge_expsum(const int* __restrict__ src, const int* __restrict__ dst,
                            int E, const float* __restrict__ as,
                            const float* __restrict__ ad,
                            const float* __restrict__ m, float* __restrict__ den,
                            int total) {
  int idx = blockIdx.x * blockDim.x + threadIdx.x;
  if (idx >= total) return;
  int s, d, hh;
  edge_sd<H>(idx, src, dst, E, &s, &d, &hh);
  float ev = leaky(as[s * H + hh] + ad[d * H + hh]);
  atomicAdd(&den[d * H + hh], __expf(ev - m[d * H + hh]));
}

template <int H, int C>
__global__ void edge_agg(const int* __restrict__ src, const int* __restrict__ dst,
                         int E, const float* __restrict__ as,
                         const float* __restrict__ ad,
                         const float* __restrict__ m,
                         const float* __restrict__ den,
                         const float* __restrict__ h, float* __restrict__ out,
                         int total) {
  int idx = blockIdx.x * blockDim.x + threadIdx.x;
  if (idx >= total) return;
  int s, d, hh;
  edge_sd<H>(idx, src, dst, E, &s, &d, &hh);
  int dh = d * H + hh;
  float ev = leaky(as[s * H + hh] + ad[dh]);
  float w  = __expf(ev - m[dh]) / (den[dh] + 1e-16f);
  const float4* hs = (const float4*)(h + ((size_t)s * H + hh) * C);
  float* op = out + ((size_t)d * H + hh) * C;
#pragma unroll
  for (int c = 0; c < C / 4; ++c) {
    float4 v = hs[c];                       // global_load_b128 gather
    atomicAdd(&op[c * 4 + 0], w * v.x);     // global_atomic_add_f32 scatter
    atomicAdd(&op[c * 4 + 1], w * v.y);
    atomicAdd(&op[c * 4 + 2], w * v.z);
    atomicAdd(&op[c * 4 + 3], w * v.w);
  }
}

__global__ void bias_relu(float* __restrict__ v, const float* __restrict__ b,
                          int D, int total) {
  int idx = blockIdx.x * blockDim.x + threadIdx.x;
  if (idx >= total) return;
  v[idx] = fmaxf(v[idx] + b[idx % D], 0.f);
}

// one wave32 per node: lane == channel (C==32); log_softmax via xor-shuffles
__global__ void finalize_logsoftmax(const float* __restrict__ acc,
                                    const float* __restrict__ b,
                                    float* __restrict__ out, int n_nodes) {
  int gid = blockIdx.x * blockDim.x + threadIdx.x;
  int n = gid >> 5;
  int c = gid & 31;
  if (n >= n_nodes) return;
  float v = acc[(size_t)n * 32 + c] + b[c];
  float mx = v;
#pragma unroll
  for (int off = 16; off; off >>= 1) mx = fmaxf(mx, __shfl_xor(mx, off));
  float ex = __expf(v - mx), sm = ex;
#pragma unroll
  for (int off = 16; off; off >>= 1) sm += __shfl_xor(sm, off);
  out[(size_t)n * 32 + c] = (v - mx) - __logf(sm);
}

extern "C" void kernel_launch(void* const* d_in, const int* in_sizes, int n_in,
                              void* d_out, int out_size, void* d_ws, size_t ws_size,
                              hipStream_t stream) {
  const float* x    = (const float*)d_in[0];
  const int*   ei   = (const int*)d_in[1];
  const float* W1   = (const float*)d_in[2];
  const float* a_s1 = (const float*)d_in[3];
  const float* a_d1 = (const float*)d_in[4];
  const float* b1   = (const float*)d_in[5];
  const float* W2   = (const float*)d_in[6];
  const float* a_s2 = (const float*)d_in[7];
  const float* a_d2 = (const float*)d_in[8];
  const float* b2   = (const float*)d_in[9];

  const int n  = NNODES;
  const int E  = in_sizes[1] / 2;
  const int EA = E + n;                 // with self loops
  const int* src = ei;
  const int* dst = ei + E;

  // workspace carve-up (floats)
  float* ws   = (float*)d_ws;
  float* h1   = ws;                          // n*D1
  float* out1 = h1   + (size_t)n * D1;       // n*D1
  float* as1  = out1 + (size_t)n * D1;       // n*HEADS1
  float* ad1  = as1  + (size_t)n * HEADS1;
  float* m1   = ad1  + (size_t)n * HEADS1;
  float* dn1  = m1   + (size_t)n * HEADS1;
  // layer-2 buffers alias layer-1 regions that are dead by then
  float* h2   = h1;                          // n*CH2
  float* out2 = h1 + (size_t)n * CH2;        // n*CH2
  float* as2  = as1; float* ad2 = ad1; float* m2 = m1; float* dn2 = dn1;

  const int TB = 256;

  // ---------------- layer 1 ----------------
  hipMemsetAsync(out1, 0,    (size_t)n * D1 * sizeof(float), stream);
  hipMemsetAsync(dn1,  0,    (size_t)n * HEADS1 * sizeof(float), stream);
  hipMemsetAsync(m1,   0xFF, (size_t)n * HEADS1 * sizeof(float), stream);

  wmma_gemm<F_IN><<<dim3(n / 16, D1 / 32), 32, 0, stream>>>(x, W1, h1, D1);

  { int t = n * HEADS1;
    attn_coeff<HEADS1, CH1><<<(t + TB - 1) / TB, TB, 0, stream>>>(h1, a_s1, a_d1, as1, ad1, t); }
  { int t = EA * HEADS1;
    edge_max<HEADS1>   <<<(t + TB - 1) / TB, TB, 0, stream>>>(src, dst, E, as1, ad1, m1, t);
    edge_expsum<HEADS1><<<(t + TB - 1) / TB, TB, 0, stream>>>(src, dst, E, as1, ad1, m1, dn1, t);
    edge_agg<HEADS1, CH1><<<(t + TB - 1) / TB, TB, 0, stream>>>(src, dst, E, as1, ad1, m1, dn1,
                                                                h1, out1, t); }
  { int t = n * D1;
    bias_relu<<<(t + TB - 1) / TB, TB, 0, stream>>>(out1, b1, D1, t); }

  // ---------------- layer 2 ----------------
  hipMemsetAsync(out2, 0,    (size_t)n * CH2 * sizeof(float), stream);
  hipMemsetAsync(dn2,  0,    (size_t)n * sizeof(float), stream);
  hipMemsetAsync(m2,   0xFF, (size_t)n * sizeof(float), stream);

  wmma_gemm<D1><<<dim3(n / 16, CH2 / 32), 32, 0, stream>>>(out1, W2, h2, CH2);

  { int t = n;
    attn_coeff<1, CH2><<<(t + TB - 1) / TB, TB, 0, stream>>>(h2, a_s2, a_d2, as2, ad2, t); }
  { int t = EA;
    edge_max<1>   <<<(t + TB - 1) / TB, TB, 0, stream>>>(src, dst, E, as2, ad2, m2, t);
    edge_expsum<1><<<(t + TB - 1) / TB, TB, 0, stream>>>(src, dst, E, as2, ad2, m2, dn2, t);
    edge_agg<1, CH2><<<(t + TB - 1) / TB, TB, 0, stream>>>(src, dst, E, as2, ad2, m2, dn2,
                                                           h2, out2, t); }

  { int t = n * 32;
    finalize_logsoftmax<<<(t + TB - 1) / TB, TB, 0, stream>>>(out2, b2, (float*)d_out, n); }
}
